// MultiheadAttention_38001870635043
// MI455X (gfx1250) — matrix operand order repaired and checked by reading, not compiled, not run on hardware
//
#include <hip/hip_runtime.h>

typedef __bf16 bf16_t;
typedef __attribute__((ext_vector_type(16))) __bf16 v16bf;
typedef __attribute__((ext_vector_type(8)))  __bf16 v8bf;
typedef __attribute__((ext_vector_type(4)))  __bf16 v4bf;
typedef __attribute__((ext_vector_type(8)))  float  v8f;

#define COMPILER_MEM_BARRIER() asm volatile("" ::: "memory")

constexpr int kDModel = 1024;
constexpr int kHeads  = 16;
constexpr int kDk     = 64;
constexpr int kBatch  = 2;
constexpr int kSeq    = 2048;
constexpr int kRows   = kBatch * kSeq;  // 4096

static __device__ inline v8f v8f_zero() {
  v8f v;
#pragma unroll
  for (int i = 0; i < 8; ++i) v[i] = 0.0f;
  return v;
}

static __device__ inline v16bf make_v16(v8bf lo, v8bf hi) {
  return __builtin_shufflevector(lo, hi, 0, 1, 2, 3, 4, 5, 6, 7, 8, 9, 10, 11, 12, 13, 14, 15);
}

// ---------------------------------------------------------------------------
// f32 -> bf16 conversion (vectorized by 4)
// ---------------------------------------------------------------------------
__global__ __launch_bounds__(256) void cvt_f32_bf16(const float* __restrict__ src,
                                                    bf16_t* __restrict__ dst, int n4) {
  int i = blockIdx.x * blockDim.x + threadIdx.x;
  if (i < n4) {
    float4 f = ((const float4*)src)[i];
    v4bf o;
    o[0] = (bf16_t)f.x; o[1] = (bf16_t)f.y; o[2] = (bf16_t)f.z; o[3] = (bf16_t)f.w;
    ((v4bf*)dst)[i] = o;
  }
}

// ---------------------------------------------------------------------------
// Y[M,N] = A[M,K] * W[N,K]^T   (M=4096, N=K=1024), bf16 in, f32 accum.
// OUT_MODE 0: f32 row-major   1: bf16 row-major   2: bf16 transposed-per-head
//   (mode 2: Vt[((b*H+h)*64+d)*2048 + s] so attention PV B-fragments are
//    contiguous along the key dimension)
// Block = 256 thr (8 waves). Block tile 128x128; wave tile 32x64 (2x4 WMMA).
// Register double-buffered over K so global loads overlap WMMA issue.
// ---------------------------------------------------------------------------
template <int OUT_MODE>
__global__ __launch_bounds__(256) void gemm_xwt(const bf16_t* __restrict__ A,
                                                const bf16_t* __restrict__ W,
                                                void* __restrict__ outp) {
  const int lane = threadIdx.x & 31;
  const int wave = threadIdx.x >> 5;
  const int lr = lane & 15;   // row/col within 16-wide fragment
  const int lh = lane >> 4;   // half-select (K split per ISA layout)
  const int wm = wave & 3, wn = wave >> 2;
  const int m0 = blockIdx.y * 128 + wm * 32;
  const int n0 = blockIdx.x * 128 + wn * 64;

  v8f acc[2][4];
#pragma unroll
  for (int mi = 0; mi < 2; ++mi)
#pragma unroll
    for (int ni = 0; ni < 4; ++ni) acc[mi][ni] = v8f_zero();

  // A fragment (this code's k order, B matches): lane holds row, 16 contiguous k at k=lh*16.
  const bf16_t* aB0 = A + (size_t)(m0 + lr) * kDModel + lh * 16;
  const bf16_t* aB1 = aB0 + (size_t)16 * kDModel;
  // B fragment (same k order as A): lane holds col n, 16 contiguous k at
  // k = lh*16 (lanes 0-15: K0-15, lanes 16-31: K16-31).
  const bf16_t* wB[4];
#pragma unroll
  for (int ni = 0; ni < 4; ++ni)
    wB[ni] = W + (size_t)(n0 + ni * 16 + lr) * kDModel + lh * 16;

  auto loadA = [&](int k0, v16bf (&af)[2]) {
    af[0] = make_v16(*(const v8bf*)(aB0 + k0), *(const v8bf*)(aB0 + k0 + 8));
    af[1] = make_v16(*(const v8bf*)(aB1 + k0), *(const v8bf*)(aB1 + k0 + 8));
  };
  auto loadB = [&](int k0, v16bf (&bfr)[4]) {
#pragma unroll
    for (int ni = 0; ni < 4; ++ni) bfr[ni] = *(const v16bf*)(wB[ni] + k0);
  };
  auto mmas = [&](v16bf (&af)[2], v16bf (&bfr)[4]) {
#pragma unroll
    for (int mi = 0; mi < 2; ++mi)
#pragma unroll
      for (int ni = 0; ni < 4; ++ni)
        acc[mi][ni] = __builtin_amdgcn_wmma_f32_16x16x32_bf16(
            false, af[mi], false, bfr[ni], (short)0, acc[mi][ni], false, false);
  };

  v16bf aCur[2], bCur[4], aNxt[2], bNxt[4];
  loadA(0, aCur);
  loadB(0, bCur);
  for (int k0 = 0; k0 < kDModel; k0 += 64) {
    loadA(k0 + 32, aNxt);          // issue next-step loads before consuming cur
    loadB(k0 + 32, bNxt);
    if (k0 + 128 < kDModel) {      // gfx1250 global_prefetch_b8, 2 steps ahead
      __builtin_prefetch(aB0 + k0 + 128, 0, 1);
      __builtin_prefetch(aB1 + k0 + 128, 0, 1);
#pragma unroll
      for (int ni = 0; ni < 4; ++ni) __builtin_prefetch(wB[ni] + k0 + 128, 0, 1);
    }
    mmas(aCur, bCur);
    if (k0 + 64 < kDModel) {
      loadA(k0 + 64, aCur);
      loadB(k0 + 64, bCur);
    }
    mmas(aNxt, bNxt);
  }

  // C/D layout: lane col = lr, VGPR v -> row v + 8*lh.
#pragma unroll
  for (int mi = 0; mi < 2; ++mi)
#pragma unroll
    for (int ni = 0; ni < 4; ++ni) {
      const int col = n0 + ni * 16 + lr;
#pragma unroll
      for (int v = 0; v < 8; ++v) {
        const int row = m0 + mi * 16 + lh * 8 + v;
        const float val = acc[mi][ni][v];
        if constexpr (OUT_MODE == 0) {
          ((float*)outp)[(size_t)row * kDModel + col] = val;
        } else if constexpr (OUT_MODE == 1) {
          ((bf16_t*)outp)[(size_t)row * kDModel + col] = (bf16_t)val;
        } else {
          const int b = row >> 11, s = row & (kSeq - 1);
          const int h = col >> 6, d = col & (kDk - 1);
          ((bf16_t*)outp)[((size_t)(b * kHeads + h) * kDk + d) * kSeq + s] = (bf16_t)val;
        }
      }
    }
}

// ---------------------------------------------------------------------------
// Fused causal flash attention. Grid (S/128, H, B), 256 thr = 8 waves.
// Each wave owns 16 query rows; online softmax over key blocks of 32.
// K fragments double-buffered (ping-pong, no swap movs); V fragments issued
// at step entry so they retire under the softmax; prefetch 2 steps ahead.
// ---------------------------------------------------------------------------
__global__ __launch_bounds__(256) void attn_fused(const bf16_t* __restrict__ Q,
                                                  const bf16_t* __restrict__ K,
                                                  const bf16_t* __restrict__ Vt,
                                                  bf16_t* __restrict__ Ctx) {
  __shared__ float  sbuf[8][16][33];  // wave-private score tile (f32)
  __shared__ bf16_t pbuf[8][16][40];  // wave-private P tile (16B-aligned rows)
  __shared__ float  rbuf[8][16];      // per-row broadcast (rescale / 1/l)

  const int lane = threadIdx.x & 31;
  const int wave = threadIdx.x >> 5;
  const int lr = lane & 15, lh = lane >> 4;
  const int b = blockIdx.z, h = blockIdx.y;
  const int bh = b * kHeads + h;
  const int q0 = blockIdx.x * 128 + wave * 16;
  const int bsb = b * kSeq;
  const size_t headOff = (size_t)h * kDk;

  // Q fragments: two K=32 halves of dk=64.
  v16bf qf[2];
  {
    const bf16_t* qp = Q + (size_t)(bsb + q0 + lr) * kDModel + headOff + lh * 16;
    qf[0] = make_v16(*(const v8bf*)(qp), *(const v8bf*)(qp + 8));
    qf[1] = make_v16(*(const v8bf*)(qp + 32), *(const v8bf*)(qp + 40));
  }

  v8f o[4];
#pragma unroll
  for (int nc = 0; nc < 4; ++nc) o[nc] = v8f_zero();
  float m_r = -1e30f, l_r = 0.0f;

  const int klim = q0 + 16;  // keys needed by this wave's rows (causal)

  auto loadK4 = [&](int j0, v16bf (&kf)[4]) {
#pragma unroll
    for (int jt = 0; jt < 2; ++jt) {
      const bf16_t* kp =
          K + (size_t)(bsb + j0 + jt * 16 + lr) * kDModel + headOff + lh * 16;
      kf[jt * 2 + 0] = *(const v16bf*)(kp);
      kf[jt * 2 + 1] = *(const v16bf*)(kp + 32);
    }
  };

  auto step = [&](int j0, v16bf (&kcur)[4], v16bf (&knxt)[4]) {
    // ---- V fragments: no dependence on softmax, issue first ----
    const bf16_t* vb = Vt + ((size_t)bh * kDk + lr) * kSeq + j0 + lh * 16;
    v16bf vf[4];
#pragma unroll
    for (int nc = 0; nc < 4; ++nc)
      vf[nc] = *(const v16bf*)(vb + (size_t)nc * 16 * kSeq);

    // ---- scores: S = Q * K^T for 2 key subtiles of 16 ----
    v8f st[2];
#pragma unroll
    for (int jt = 0; jt < 2; ++jt) {
      v8f s = v8f_zero();
      s = __builtin_amdgcn_wmma_f32_16x16x32_bf16(false, qf[0], false, kcur[jt * 2 + 0], (short)0, s, false, false);
      s = __builtin_amdgcn_wmma_f32_16x16x32_bf16(false, qf[1], false, kcur[jt * 2 + 1], (short)0, s, false, false);
      st[jt] = s;
    }

    // ---- issue next step's K loads; they retire under the softmax ----
    if (j0 + 32 < klim) loadK4(j0 + 32, knxt);
    if (j0 + 64 < klim) {  // gfx1250 global_prefetch_b8, 2 steps ahead
      __builtin_prefetch(K + (size_t)(bsb + j0 + 64 + lane) * kDModel + headOff, 0, 1);
      __builtin_prefetch(Vt + ((size_t)bh * kDk + lane) * kSeq + j0 + 64, 0, 1);
      __builtin_prefetch(Vt + ((size_t)bh * kDk + 32 + lane) * kSeq + j0 + 64, 0, 1);
    }

    // spill scores (pre-scaled by 1/sqrt(dk)) to wave-private LDS
#pragma unroll
    for (int jt = 0; jt < 2; ++jt)
#pragma unroll
      for (int v = 0; v < 8; ++v)
        sbuf[wave][lh * 8 + v][jt * 16 + lr] = st[jt][v] * 0.125f;
    COMPILER_MEM_BARRIER();  // DS ops are in-order within a wave

    // ---- online softmax: lanes 0-15 each own one row ----
    if (lh == 0) {
      const int r = lr, qg = q0 + r;
      float mx = m_r;
#pragma unroll
      for (int c = 0; c < 32; ++c)
        if (j0 + c <= qg) mx = fmaxf(mx, sbuf[wave][r][c]);
      const float sc = __expf(m_r - mx);
      float rs = 0.0f;
#pragma unroll
      for (int c = 0; c < 32; ++c) {
        const float p = (j0 + c <= qg) ? __expf(sbuf[wave][r][c] - mx) : 0.0f;
        rs += p;
        pbuf[wave][r][c] = (bf16_t)p;
      }
      l_r = l_r * sc + rs;
      m_r = mx;
      rbuf[wave][r] = sc;
    }
    COMPILER_MEM_BARRIER();

    // rescale accumulator rows by exp(m_old - m_new)
    float scv[8];
#pragma unroll
    for (int v = 0; v < 8; ++v) scv[v] = rbuf[wave][lh * 8 + v];
#pragma unroll
    for (int nc = 0; nc < 4; ++nc)
#pragma unroll
      for (int v = 0; v < 8; ++v) o[nc][v] *= scv[v];

    // ---- O += P * V (V fragments loaded long ago) ----
    v16bf pf;
    {
      const bf16_t* pp = &pbuf[wave][lr][lh * 16];
      pf = make_v16(*(const v8bf*)(pp), *(const v8bf*)(pp + 8));
    }
#pragma unroll
    for (int nc = 0; nc < 4; ++nc)
      o[nc] = __builtin_amdgcn_wmma_f32_16x16x32_bf16(false, pf, false, vf[nc], (short)0, o[nc], false, false);
  };

  v16bf kA[4], kB[4];
  loadK4(0, kA);
  int j0 = 0;
  while (true) {
    step(j0, kA, kB);
    j0 += 32;
    if (j0 >= klim) break;
    step(j0, kB, kA);
    j0 += 32;
    if (j0 >= klim) break;
  }

  // ---- finalize: divide by row sums, write bf16 context ----
  if (lh == 0) rbuf[wave][lr] = 1.0f / l_r;
  COMPILER_MEM_BARRIER();
  float inv[8];
#pragma unroll
  for (int v = 0; v < 8; ++v) inv[v] = rbuf[wave][lh * 8 + v];
#pragma unroll
  for (int nc = 0; nc < 4; ++nc)
#pragma unroll
    for (int v = 0; v < 8; ++v)
      Ctx[(size_t)(bsb + q0 + lh * 8 + v) * kDModel + headOff + nc * 16 + lr] =
          (bf16_t)(o[nc][v] * inv[v]);
}

// ---------------------------------------------------------------------------
extern "C" void kernel_launch(void* const* d_in, const int* in_sizes, int n_in,
                              void* d_out, int out_size, void* d_ws, size_t ws_size,
                              hipStream_t stream) {
  (void)in_sizes; (void)n_in; (void)out_size; (void)ws_size;
  const float* x  = (const float*)d_in[0];
  const float* Wq = (const float*)d_in[1];
  const float* Wk = (const float*)d_in[2];
  const float* Wv = (const float*)d_in[3];
  const float* Wo = (const float*)d_in[4];
  float* out = (float*)d_out;

  bf16_t* w = (bf16_t*)d_ws;
  bf16_t* xb  = w; w += (size_t)kRows * kDModel;
  bf16_t* wqb = w; w += (size_t)kDModel * kDModel;
  bf16_t* wkb = w; w += (size_t)kDModel * kDModel;
  bf16_t* wvb = w; w += (size_t)kDModel * kDModel;
  bf16_t* wob = w; w += (size_t)kDModel * kDModel;
  bf16_t* Qb  = w; w += (size_t)kRows * kDModel;
  bf16_t* Kb  = w; w += (size_t)kRows * kDModel;
  bf16_t* Vtb = w; w += (size_t)kRows * kDModel;
  bf16_t* Ctx = w; w += (size_t)kRows * kDModel;

  const int n4x = kRows * kDModel / 4;      // 1,048,576
  const int n4w = kDModel * kDModel / 4;    //   262,144
  cvt_f32_bf16<<<n4x / 256, 256, 0, stream>>>(x, xb, n4x);
  cvt_f32_bf16<<<n4w / 256, 256, 0, stream>>>(Wq, wqb, n4w);
  cvt_f32_bf16<<<n4w / 256, 256, 0, stream>>>(Wk, wkb, n4w);
  cvt_f32_bf16<<<n4w / 256, 256, 0, stream>>>(Wv, wvb, n4w);
  cvt_f32_bf16<<<n4w / 256, 256, 0, stream>>>(Wo, wob, n4w);

  dim3 gGrid(kDModel / 128, kRows / 128);   // (8, 32)
  gemm_xwt<1><<<gGrid, 256, 0, stream>>>(xb, wqb, (void*)Qb);
  gemm_xwt<1><<<gGrid, 256, 0, stream>>>(xb, wkb, (void*)Kb);
  gemm_xwt<2><<<gGrid, 256, 0, stream>>>(xb, wvb, (void*)Vtb);

  attn_fused<<<dim3(kSeq / 128, kHeads, kBatch), 256, 0, stream>>>(Qb, Kb, Vtb, Ctx);

  gemm_xwt<0><<<gGrid, 256, 0, stream>>>(Ctx, wob, (void*)out);
}
